// GraphSAGE_46042049413864
// MI455X (gfx1250) — matrix operand order repaired and checked by reading, not compiled, not run on hardware
//
#include <hip/hip_runtime.h>

// GraphSAGE (mean) x4 on MI455X / gfx1250.
//
// Roofline: aggregation dominates (E=1.6M edges x 256B bf16 rows = 0.41 GB/layer
// after moving h to bf16 + CSR gather; was 1.65 GB/layer with fp32 scatter
// atomics). GEMMs are ~22 GFLOP total -> v_wmma_f32_16x16x32_bf16 with
// LDS-staged fused weights. CSR is built once and reused by all 4 layers.

typedef __attribute__((ext_vector_type(16))) __bf16 v16bf;
typedef __attribute__((ext_vector_type(8)))  __bf16 v8bf;
typedef __attribute__((ext_vector_type(4)))  __bf16 v4bf;
typedef __attribute__((ext_vector_type(8)))  float  v8f;

// ---------------------------------------------------------------- utilities

__global__ void fill_i32_kernel(int* __restrict__ p, int v, int n) {
    int i = blockIdx.x * blockDim.x + threadIdx.x;
    const int stride = gridDim.x * blockDim.x;
    for (; i < n; i += stride) p[i] = v;
}

__global__ void cvt_f32_bf16_kernel(const float* __restrict__ in, __bf16* __restrict__ out, long n) {
    // n multiple of 4 (N*128)
    const long n4 = n >> 2;
    const long stride = (long)gridDim.x * blockDim.x;
    for (long i = (long)blockIdx.x * blockDim.x + threadIdx.x; i < n4; i += stride) {
        const float4 f = ((const float4*)in)[i];
        v4bf o;
        o[0] = (__bf16)f.x; o[1] = (__bf16)f.y; o[2] = (__bf16)f.z; o[3] = (__bf16)f.w;
        *(v4bf*)(out + i * 4) = o;
    }
}

// Fuse w_self (out x 128) and w_neigh (out x 128) into bf16 Wcat (npad x 256),
// zero-padded rows for npad > nout (layer 4: 40 -> 48).
__global__ void conv_w_kernel(const float* __restrict__ wsl, const float* __restrict__ wng,
                              __bf16* __restrict__ out, int nout, int npad) {
    int i = blockIdx.x * blockDim.x + threadIdx.x;
    if (i >= npad * 256) return;
    int o = i >> 8;
    int k = i & 255;
    float v = 0.0f;
    if (o < nout) v = (k < 128) ? wsl[o * 128 + k] : wng[o * 128 + (k - 128)];
    out[i] = (__bf16)v;
}

// -------------------------------------------------------------- CSR build

__global__ void hist_kernel(const int* __restrict__ dst, int* __restrict__ cnt, int ne) {
    int e = blockIdx.x * blockDim.x + threadIdx.x;
    if (e < ne) atomicAdd(&cnt[dst[e]], 1);
}

// Single-workgroup exclusive scan: rowptr[0]=0, rowptr[i+1]=sum(cnt[0..i]).
// One-time cost, 1024 threads, Hillis-Steele per 1024-chunk + running offset.
__global__ __launch_bounds__(1024)
void scan_kernel(const int* __restrict__ cnt, int* __restrict__ rowptr, int n) {
    __shared__ int sbuf[1024];
    __shared__ int soff;
    const int tid = threadIdx.x;
    if (tid == 0) { soff = 0; rowptr[0] = 0; }
    __syncthreads();
    for (int base = 0; base < n; base += 1024) {
        const int i = base + tid;
        sbuf[tid] = (i < n) ? cnt[i] : 0;
        __syncthreads();
#pragma unroll
        for (int d = 1; d < 1024; d <<= 1) {
            const int t = (tid >= d) ? sbuf[tid - d] : 0;
            __syncthreads();
            sbuf[tid] += t;
            __syncthreads();
        }
        if (i < n) rowptr[i + 1] = sbuf[tid] + soff;
        __syncthreads();
        if (tid == 1023) soff += sbuf[1023];
        __syncthreads();
    }
}

__global__ void place_kernel(const int* __restrict__ src, const int* __restrict__ dst,
                             const int* __restrict__ rowptr, int* __restrict__ cnt,
                             int* __restrict__ colidx, int ne) {
    int e = blockIdx.x * blockDim.x + threadIdx.x;
    if (e < ne) {
        const int d = dst[e];
        const int pos = rowptr[d] + atomicAdd(&cnt[d], 1);
        colidx[pos] = src[e];
    }
}

// ----------------------------------------------------------- gather-mean
// One wave per node: walk in-edges via CSR, each lane owns 4 features
// (8B bf16 loads, coalesced 256B per neighbor row), accumulate fp32 in
// registers, scale by 1/deg, write one bf16 agg row. Prefetch next row.
__global__ __launch_bounds__(256)
void gather_mean_kernel(const __bf16* __restrict__ h,
                        const int* __restrict__ rowptr,
                        const int* __restrict__ colidx,
                        __bf16* __restrict__ aggv,
                        int nnodes) {
    const int lane = threadIdx.x & 31;
    const int wv   = (int)((blockIdx.x * blockDim.x + threadIdx.x) >> 5);
    const int nw   = (int)((gridDim.x * blockDim.x) >> 5);
    for (int v = wv; v < nnodes; v += nw) {
        const int s0 = rowptr[v];
        const int s1 = rowptr[v + 1];
        float a0 = 0.0f, a1 = 0.0f, a2 = 0.0f, a3 = 0.0f;
        for (int j = s0; j < s1; ++j) {
            const int s = colidx[j];
            if (j + 1 < s1)
                __builtin_prefetch(h + (size_t)colidx[j + 1] * 128 + lane * 4, 0, 1);
            const v4bf hv = *(const v4bf*)(h + (size_t)s * 128 + lane * 4);
            a0 += (float)hv[0]; a1 += (float)hv[1];
            a2 += (float)hv[2]; a3 += (float)hv[3];
        }
        const float sc = 1.0f / (float)max(s1 - s0, 1);
        v4bf o;
        o[0] = (__bf16)(a0 * sc); o[1] = (__bf16)(a1 * sc);
        o[2] = (__bf16)(a2 * sc); o[3] = (__bf16)(a3 * sc);
        *(v4bf*)(aggv + (size_t)v * 128 + lane * 4) = o;
    }
}

// ------------------------------------------------------------- WMMA GEMM
// hout[r, c] = act( sum_k [h | agg][r, k] * Wcat[c, k] + bias[c] )   (agg pre-scaled)
// Block = 256 thr (8 waves). Block stages Wcat (bf16, NTILE*16 x 256) in LDS.
// Each wave: 16 rows x NTILE*16 cols, K=256 in 8 steps of 32 (bf16 WMMA).
// Row index CLAMPED for loads; stores predicated.
//
// VGPR layouts used here (A/B k order contiguous, not the ISA table order):
//  A (16x32 bf16): lane L -> m = L%16, half = L/16;
//                  element i: K = half*16 + i
//  B (32x16 bf16): lane L -> n = L%16; element i: K = (L/16)*16 + i
//  C (16x16 f32) : lane L -> n = L%16; vgpr v: M = v + 8*(L/16)
template <int NTILE, int NCOLS, bool RELU, typename OUT>
__global__ __launch_bounds__(256)
void sage_gemm_kernel(const __bf16* __restrict__ hin,
                      const __bf16* __restrict__ aggv,
                      const __bf16* __restrict__ wcat,
                      const float* __restrict__ bias,
                      OUT* __restrict__ hout,
                      int nrows) {
    __shared__ __bf16 sW[NTILE * 16 * 256];   // 64 KB (NTILE=8) / 24 KB (NTILE=3)

    const int tid = threadIdx.x;
    {   // cooperative stage of weights into LDS (16B vectors)
        const uint4* g = (const uint4*)wcat;
        uint4* s = (uint4*)sW;
        const int nvec = NTILE * 16 * 32;
        for (int i = tid; i < nvec; i += 256) s[i] = g[i];
    }
    __syncthreads();

    const int wave = tid >> 5;
    const int lane = tid & 31;
    const int m    = lane & 15;
    const int half = lane >> 4;

    const int row0 = (blockIdx.x * 8 + wave) * 16;
    if (row0 >= nrows) return;                       // wave-uniform exit
    const int rowc = min(row0 + m, nrows - 1);       // clamped -> loads always valid

    const __bf16* hrow = hin  + (size_t)rowc * 128;
    const __bf16* arow = aggv + (size_t)rowc * 128;

    v8f acc[NTILE];
#pragma unroll
    for (int t = 0; t < NTILE; ++t)
#pragma unroll
        for (int e = 0; e < 8; ++e) acc[t][e] = 0.0f;

#pragma unroll
    for (int kk = 0; kk < 8; ++kk) {
        const int k0 = kk * 32;
        const __bf16* base = (k0 < 128) ? hrow : arow;
        const int     koff = (k0 < 128) ? k0 : (k0 - 128);

        // two contiguous 8xbf16 (16B) runs per lane -> global_load_b128
        const v8bf r0 = *(const v8bf*)(base + koff + half * 16);
        const v8bf r1 = *(const v8bf*)(base + koff + half * 16 + 8);
        const v16bf a = __builtin_shufflevector(r0, r1,
            0, 1, 2, 3, 4, 5, 6, 7, 8, 9, 10, 11, 12, 13, 14, 15);

#pragma unroll
        for (int t = 0; t < NTILE; ++t) {
            const v16bf b =
                *(const v16bf*)(sW + ((size_t)(t * 16 + m) * 256 + k0 + half * 16));
            acc[t] = __builtin_amdgcn_wmma_f32_16x16x32_bf16(
                false, a, false, b, (short)0, acc[t], false, false);
        }
    }

    // epilogue: bias + optional ReLU, predicated stores (bf16 or f32)
#pragma unroll
    for (int t = 0; t < NTILE; ++t) {
        const int  col = t * 16 + m;
        const bool cv  = col < NCOLS;
        const float bv = cv ? bias[col] : 0.0f;
#pragma unroll
        for (int v = 0; v < 8; ++v) {
            const int r = row0 + half * 8 + v;
            if (cv && r < nrows) {
                float val = acc[t][v] + bv;
                if (RELU) val = fmaxf(val, 0.0f);
                hout[(size_t)r * NCOLS + col] = (OUT)val;
            }
        }
    }
}

// --------------------------------------------------------------- launcher

extern "C" void kernel_launch(void* const* d_in, const int* in_sizes, int n_in,
                              void* d_out, int out_size, void* d_ws, size_t ws_size,
                              hipStream_t stream) {
    const float* x   = (const float*)d_in[0];
    const int*   src = (const int*)d_in[1];
    const int*   dst = (const int*)d_in[2];
    const int nn = in_sizes[0] / 128;   // 100000 nodes
    const int ne = in_sizes[1];         // 1600000 edges

    const float *wsl[4], *wng[4], *bs[4];
    for (int l = 0; l < 4; ++l) {
        wsl[l] = (const float*)d_in[3 + 3 * l];
        wng[l] = (const float*)d_in[4 + 3 * l];
        bs[l]  = (const float*)d_in[5 + 3 * l];
    }

    // workspace carve-up (~110 MB)
    char* w = (char*)d_ws;
    auto carve = [&](size_t bytes) -> char* {
        char* p = w;
        w += (bytes + 255) & ~(size_t)255;
        return p;
    };
    int* rowptr = (int*)carve((size_t)(nn + 1) * 4);
    int* cnt    = (int*)carve((size_t)nn * 4);
    int* colidx = (int*)carve((size_t)ne * 4);
    const size_t featB = (size_t)nn * 128 * sizeof(__bf16);
    __bf16* h0   = (__bf16*)carve(featB);
    __bf16* h1   = (__bf16*)carve(featB);
    __bf16* h2   = (__bf16*)carve(featB);
    __bf16* aggv = (__bf16*)carve(featB);
    __bf16* wc[4];
    for (int l = 0; l < 4; ++l) wc[l] = (__bf16*)carve((size_t)128 * 256 * sizeof(__bf16));

    // ---- CSR build (once, reused by all 4 layers)
    fill_i32_kernel<<<256, 256, 0, stream>>>(cnt, 0, nn);
    hist_kernel<<<(ne + 255) / 256, 256, 0, stream>>>(dst, cnt, ne);
    scan_kernel<<<1, 1024, 0, stream>>>(cnt, rowptr, nn);
    fill_i32_kernel<<<256, 256, 0, stream>>>(cnt, 0, nn);
    place_kernel<<<(ne + 255) / 256, 256, 0, stream>>>(src, dst, rowptr, cnt, colidx, ne);

    // ---- bf16 input features + fused bf16 weights
    cvt_f32_bf16_kernel<<<2048, 256, 0, stream>>>(x, h0, (long)nn * 128);
    const int npads[4] = {128, 128, 128, 48};
    const int nouts[4] = {128, 128, 128, 40};
    for (int l = 0; l < 4; ++l)
        conv_w_kernel<<<(npads[l] * 256 + 255) / 256, 256, 0, stream>>>(
            wsl[l], wng[l], wc[l], nouts[l], npads[l]);

    // ---- 4 SAGE layers
    const __bf16* hin = h0;
    const int gemm_blocks = (nn + 127) / 128;    // 128 rows per block (8 waves x 16)
    __bf16* houts_bf[3] = {h1, h2, h1};
    for (int l = 0; l < 4; ++l) {
        gather_mean_kernel<<<4096, 256, 0, stream>>>(hin, rowptr, colidx, aggv, nn);
        if (l < 3) {
            sage_gemm_kernel<8, 128, true, __bf16><<<gemm_blocks, 256, 0, stream>>>(
                hin, aggv, wc[l], bs[l], houts_bf[l], nn);
            hin = houts_bf[l];
        } else {
            sage_gemm_kernel<3, 40, false, float><<<gemm_blocks, 256, 0, stream>>>(
                hin, aggv, wc[l], bs[l], (float*)d_out, nn);
        }
    }
}
